// MultiHeadAttention_40913858462242
// MI455X (gfx1250) — matrix operand order repaired and checked by reading, not compiled, not run on hardware
//
#include <hip/hip_runtime.h>
#include <hip/hip_bf16.h>
#include <math.h>

// ---- problem constants (match reference) ----
constexpr int Bc = 2;
constexpr int Sc = 2048;
constexpr int Dc = 1024;
constexpr int Hc = 16;
constexpr int DKc = Dc / Hc;       // 64
constexpr int HDc = Hc * DKc;      // 1024

typedef __attribute__((ext_vector_type(16))) _Float16 v16h;
typedef __attribute__((ext_vector_type(8)))  float    v8f;

// ---------------------------------------------------------------------------
// WMMA operand loaders (f32 memory -> f16 registers), layouts per ISA 7.12.2
// ---------------------------------------------------------------------------

// A-matrix 16x32 (MxK), row-major source with leading dim lda.
// lane 0-15: M=lane,    K = {0..7, 16..23}
// lane16-31: M=lane-16, K = {8..15, 24..31}
__device__ __forceinline__ v16h load_A_f16(const float* __restrict__ A, int lda,
                                           int m0, int k0, int lane) {
  const int M  = lane & 15;
  const int hh = lane >> 4;
  const float* row = A + (size_t)(m0 + M) * lda + k0;
  v16h a;
#pragma unroll
  for (int j = 0; j < 8; ++j) {
    const int kb = (j < 4) ? (2 * j + 8 * hh) : (16 + 2 * (j - 4) + 8 * hh);
    a[2 * j]     = (_Float16)row[kb];
    a[2 * j + 1] = (_Float16)row[kb + 1];
  }
  return a;
}

// B-matrix 32x16 (KxN), row-major source (K rows, leading dim ldb).
// lane L holds column N = n0 + L%16; slots e are K = {0..7,16..23} (+8 for lanes 16-31).
__device__ __forceinline__ v16h load_B_f16(const float* __restrict__ Bm, int ldb,
                                           int k0, int n0, int lane) {
  const int hh = lane >> 4;
  const float* col = Bm + (size_t)k0 * ldb + n0 + (lane & 15);
  v16h b;
#pragma unroll
  for (int e = 0; e < 16; ++e) b[e] = (_Float16)col[(size_t)(e + 8 * hh + (e >= 8 ? 8 : 0)) * ldb];
  return b;
}

// B-matrix 32x16 where memory holds the transpose: B[k][n] = Bt[n][k]
// (used for Q*K^T: Bt = K-projection, row-major [S,DK]).
__device__ __forceinline__ v16h load_Bt_f16(const float* __restrict__ Bt, int ldt,
                                            int k0, int n0, int lane) {
  const int hh = lane >> 4;
  const float* row = Bt + (size_t)(n0 + (lane & 15)) * ldt + k0;
  v16h b;
#pragma unroll
  for (int e = 0; e < 16; ++e) b[e] = (_Float16)row[e + 8 * hh + (e >= 8 ? 8 : 0)];
  return b;
}

#define WMMA_F16(a, b, c) \
  __builtin_amdgcn_wmma_f32_16x16x32_f16(false, (a), false, (b), (short)0, (c), false, false)

// ---------------------------------------------------------------------------
// 1) per-head projection: Y[b,h,s,n] = sum_d X[b,s,d]*W[h,d,n] + bias[h,n]
//    One wave owns a 16x64 strip (all of DK): 4 WMMAs per K-chunk off 1 A load.
//    grid: (S/128, 1, B*H), block 256 (8 waves)
// ---------------------------------------------------------------------------
__global__ __launch_bounds__(256) void proj_kernel(const float* __restrict__ X,
                                                   const float* __restrict__ W,
                                                   const float* __restrict__ bias,
                                                   float* __restrict__ Y) {
  const int lane = threadIdx.x & 31;
  const int wave = threadIdx.x >> 5;
  const int b = blockIdx.z / Hc, h = blockIdx.z % Hc;
  const int m0 = (blockIdx.x * 8 + wave) * 16;

  const float* A  = X + (size_t)b * Sc * Dc;
  const float* Wh = W + (size_t)h * Dc * DKc;
  const float* Arow = A + (size_t)(m0 + (lane & 15)) * Dc;

  v8f c0 = {}, c1 = {}, c2 = {}, c3 = {};
  for (int k0 = 0; k0 < Dc; k0 += 32) {
    if (k0 + 32 < Dc) __builtin_prefetch(Arow + k0 + 64, 0, 1);  // global_prefetch_b8
    v16h a  = load_A_f16(A, Dc, m0, k0, lane);
    v16h b0 = load_B_f16(Wh, DKc, k0, 0, lane);
    v16h b1 = load_B_f16(Wh, DKc, k0, 16, lane);
    v16h b2 = load_B_f16(Wh, DKc, k0, 32, lane);
    v16h b3 = load_B_f16(Wh, DKc, k0, 48, lane);
    c0 = WMMA_F16(a, b0, c0);
    c1 = WMMA_F16(a, b1, c1);
    c2 = WMMA_F16(a, b2, c2);
    c3 = WMMA_F16(a, b3, c3);
  }
  const int N  = lane & 15;
  const int hh = lane >> 4;
  float* Yb = Y + ((size_t)b * Hc + h) * Sc * DKc;
  v8f acc[4] = {c0, c1, c2, c3};
#pragma unroll
  for (int t = 0; t < 4; ++t) {
    const float bvv = bias[h * DKc + t * 16 + N];
#pragma unroll
    for (int r = 0; r < 8; ++r) {
      const int M = r + 8 * hh;
      Yb[(size_t)(m0 + M) * DKc + t * 16 + N] = acc[t][r] + bvv;
    }
  }
}

// ---------------------------------------------------------------------------
// 2) scores: S[h,b,s,t] = scale * sum_k Q[b,h,s,k]*K[b,h,t,k], masked.
//    One wave owns a 16x64 strip of the SxS matrix; K=64 fully unrolled:
//    2 chunks x 4 tiles = 8 back-to-back WMMAs.
//    grid: (S/128, S/64, B*H), block 256
// ---------------------------------------------------------------------------
__global__ __launch_bounds__(256) void scores_kernel(const float* __restrict__ Q,
                                                     const float* __restrict__ Km,
                                                     const int* __restrict__ mask,
                                                     float* __restrict__ Wt) {
  const int lane = threadIdx.x & 31;
  const int wave = threadIdx.x >> 5;
  const int b = blockIdx.z / Hc, h = blockIdx.z % Hc;
  const int m0 = (blockIdx.x * 8 + wave) * 16;
  const int n0 = blockIdx.y * 64;

  const float* Qb = Q  + ((size_t)b * Hc + h) * Sc * DKc;
  const float* Kb = Km + ((size_t)b * Hc + h) * Sc * DKc;

  v8f c0 = {}, c1 = {}, c2 = {}, c3 = {};
#pragma unroll
  for (int k0 = 0; k0 < DKc; k0 += 32) {
    v16h a  = load_A_f16(Qb, DKc, m0, k0, lane);
    v16h b0 = load_Bt_f16(Kb, DKc, k0, n0 + 0, lane);   // B[k][t] = K[t][k]
    v16h b1 = load_Bt_f16(Kb, DKc, k0, n0 + 16, lane);
    v16h b2 = load_Bt_f16(Kb, DKc, k0, n0 + 32, lane);
    v16h b3 = load_Bt_f16(Kb, DKc, k0, n0 + 48, lane);
    c0 = WMMA_F16(a, b0, c0);
    c1 = WMMA_F16(a, b1, c1);
    c2 = WMMA_F16(a, b2, c2);
    c3 = WMMA_F16(a, b3, c3);
  }
  const float scale = 0.125f;  // 1/sqrt(64)
  float* out = Wt + ((size_t)h * Bc + b) * Sc * Sc;
  const int* mrow = mask + (size_t)b * Sc * Sc;
  const int N  = lane & 15;
  const int hh = lane >> 4;
  v8f acc[4] = {c0, c1, c2, c3};
#pragma unroll
  for (int t = 0; t < 4; ++t) {
#pragma unroll
    for (int r = 0; r < 8; ++r) {
      const int M = r + 8 * hh;
      const size_t idx = (size_t)(m0 + M) * Sc + n0 + t * 16 + N;
      float v = acc[t][r] * scale;
      if (mrow[idx] == 0) v = -1e9f;
      out[idx] = v;
    }
  }
}

// ---------------------------------------------------------------------------
// 3) in-place row softmax over last axis (row length S=2048), 1 block/row
// ---------------------------------------------------------------------------
__global__ __launch_bounds__(256) void softmax_kernel(float* __restrict__ Wt) {
  float* p = Wt + (size_t)blockIdx.x * Sc;
  const int t = threadIdx.x;
  const int lane = t & 31, wave = t >> 5;
  __shared__ float red[8];

  float vals[8];
  float m = -INFINITY;
#pragma unroll
  for (int i = 0; i < 8; ++i) { vals[i] = p[t + i * 256]; m = fmaxf(m, vals[i]); }
#pragma unroll
  for (int off = 16; off > 0; off >>= 1) m = fmaxf(m, __shfl_xor(m, off, 32));
  if (lane == 0) red[wave] = m;
  __syncthreads();
#pragma unroll
  for (int w = 0; w < 8; ++w) m = fmaxf(m, red[w]);

  float s = 0.f;
#pragma unroll
  for (int i = 0; i < 8; ++i) { vals[i] = __expf(vals[i] - m); s += vals[i]; }
#pragma unroll
  for (int off = 16; off > 0; off >>= 1) s += __shfl_xor(s, off, 32);
  __syncthreads();
  if (lane == 0) red[wave] = s;
  __syncthreads();
  float tot = 0.f;
#pragma unroll
  for (int w = 0; w < 8; ++w) tot += red[w];
  const float inv = 1.0f / tot;
#pragma unroll
  for (int i = 0; i < 8; ++i) p[t + i * 256] = vals[i] * inv;
}

// ---------------------------------------------------------------------------
// 4) context: Ctx[b,s,h*DK+n] = sum_t P[h,b,s,t] * V[b,h,t,n]
//    One wave owns a 16x64 strip (all of DK). grid: (S/128, 1, B*H)
// ---------------------------------------------------------------------------
__global__ __launch_bounds__(256) void ctx_kernel(const float* __restrict__ Wt,
                                                  const float* __restrict__ V,
                                                  float* __restrict__ Ctx) {
  const int lane = threadIdx.x & 31;
  const int wave = threadIdx.x >> 5;
  const int b = blockIdx.z / Hc, h = blockIdx.z % Hc;
  const int m0 = (blockIdx.x * 8 + wave) * 16;

  const float* A  = Wt + ((size_t)h * Bc + b) * Sc * Sc;   // [S,S]
  const float* Bm = V  + ((size_t)b * Hc + h) * Sc * DKc;  // [S,DK]
  const float* Arow = A + (size_t)(m0 + (lane & 15)) * Sc;

  v8f c0 = {}, c1 = {}, c2 = {}, c3 = {};
  for (int k0 = 0; k0 < Sc; k0 += 32) {
    if (k0 + 32 < Sc) __builtin_prefetch(Arow + k0 + 64, 0, 1);
    v16h a  = load_A_f16(A, Sc, m0, k0, lane);
    v16h b0 = load_B_f16(Bm, DKc, k0, 0, lane);
    v16h b1 = load_B_f16(Bm, DKc, k0, 16, lane);
    v16h b2 = load_B_f16(Bm, DKc, k0, 32, lane);
    v16h b3 = load_B_f16(Bm, DKc, k0, 48, lane);
    c0 = WMMA_F16(a, b0, c0);
    c1 = WMMA_F16(a, b1, c1);
    c2 = WMMA_F16(a, b2, c2);
    c3 = WMMA_F16(a, b3, c3);
  }
  const int N  = lane & 15;
  const int hh = lane >> 4;
  v8f acc[4] = {c0, c1, c2, c3};
#pragma unroll
  for (int t = 0; t < 4; ++t) {
#pragma unroll
    for (int r = 0; r < 8; ++r) {
      const int M = r + 8 * hh;
      Ctx[((size_t)b * Sc + m0 + M) * HDc + h * DKc + t * 16 + N] = acc[t][r];
    }
  }
}

// ---------------------------------------------------------------------------
// 5) output projection: Out[m,n] = sum_k Ctx[m,k]*Wo[k,n] + bo[n]
//    One wave owns a 16x64 strip. grid: (B*S/128, D/64, 1)
// ---------------------------------------------------------------------------
__global__ __launch_bounds__(256) void outproj_kernel(const float* __restrict__ Ctx,
                                                      const float* __restrict__ Wo,
                                                      const float* __restrict__ bo,
                                                      float* __restrict__ Out) {
  const int lane = threadIdx.x & 31;
  const int wave = threadIdx.x >> 5;
  const int m0 = (blockIdx.x * 8 + wave) * 16;
  const int n0 = blockIdx.y * 64;
  const float* Arow = Ctx + (size_t)(m0 + (lane & 15)) * HDc;

  v8f c0 = {}, c1 = {}, c2 = {}, c3 = {};
  for (int k0 = 0; k0 < HDc; k0 += 32) {
    if (k0 + 32 < HDc) __builtin_prefetch(Arow + k0 + 64, 0, 1);
    v16h a  = load_A_f16(Ctx, HDc, m0, k0, lane);
    v16h b0 = load_B_f16(Wo, Dc, k0, n0 + 0, lane);
    v16h b1 = load_B_f16(Wo, Dc, k0, n0 + 16, lane);
    v16h b2 = load_B_f16(Wo, Dc, k0, n0 + 32, lane);
    v16h b3 = load_B_f16(Wo, Dc, k0, n0 + 48, lane);
    c0 = WMMA_F16(a, b0, c0);
    c1 = WMMA_F16(a, b1, c1);
    c2 = WMMA_F16(a, b2, c2);
    c3 = WMMA_F16(a, b3, c3);
  }
  const int N  = lane & 15;
  const int hh = lane >> 4;
  v8f acc[4] = {c0, c1, c2, c3};
#pragma unroll
  for (int t = 0; t < 4; ++t) {
    const float bvv = bo[n0 + t * 16 + N];
#pragma unroll
    for (int r = 0; r < 8; ++r) {
      const int M = r + 8 * hh;
      Out[(size_t)(m0 + M) * Dc + n0 + t * 16 + N] = acc[t][r] + bvv;
    }
  }
}

// ---------------------------------------------------------------------------
// launch
// ---------------------------------------------------------------------------
extern "C" void kernel_launch(void* const* d_in, const int* in_sizes, int n_in,
                              void* d_out, int out_size, void* d_ws, size_t ws_size,
                              hipStream_t stream) {
  // setup_inputs order: k, q, v, mask, Wq, bq, Wk, bk, Wv, bv, Wo, bo
  const float* k_in  = (const float*)d_in[0];
  const float* q_in  = (const float*)d_in[1];
  const float* v_in  = (const float*)d_in[2];
  const int*   mask  = (const int*)  d_in[3];
  const float* Wq    = (const float*)d_in[4];
  const float* bq    = (const float*)d_in[5];
  const float* Wk    = (const float*)d_in[6];
  const float* bk    = (const float*)d_in[7];
  const float* Wv    = (const float*)d_in[8];
  const float* bv    = (const float*)d_in[9];
  const float* Wo    = (const float*)d_in[10];
  const float* bo    = (const float*)d_in[11];

  float* out_main = (float*)d_out;                         // [B,S,D]
  float* weights  = out_main + (size_t)Bc * Sc * Dc;       // [H,B,S,S]

  // workspace: Q, K, V ([B,H,S,DK] each) + Ctx ([B,S,H*DK])  -> 64 MB total
  const size_t projN = (size_t)Bc * Hc * Sc * DKc;
  float* Qb  = (float*)d_ws;
  float* Kb  = Qb + projN;
  float* Vb  = Kb + projN;
  float* Ctx = Vb + projN;

  const dim3 blk(256);
  const dim3 gProj(Sc / 128, 1, Bc * Hc);
  hipLaunchKernelGGL(proj_kernel, gProj, blk, 0, stream, q_in, Wq, bq, Qb);
  hipLaunchKernelGGL(proj_kernel, gProj, blk, 0, stream, k_in, Wk, bk, Kb);
  hipLaunchKernelGGL(proj_kernel, gProj, blk, 0, stream, v_in, Wv, bv, Vb);

  const dim3 gScore(Sc / 128, Sc / 64, Bc * Hc);
  hipLaunchKernelGGL(scores_kernel, gScore, blk, 0, stream, Qb, Kb, mask, weights);

  const dim3 gSoft(Hc * Bc * Sc);
  hipLaunchKernelGGL(softmax_kernel, gSoft, blk, 0, stream, weights);

  const dim3 gCtx(Sc / 128, 1, Bc * Hc);
  hipLaunchKernelGGL(ctx_kernel, gCtx, blk, 0, stream, weights, Vb, Ctx);

  const dim3 gOut((Bc * Sc) / 128, Dc / 64, 1);
  hipLaunchKernelGGL(outproj_kernel, gOut, blk, 0, stream, Ctx, Wo, bo, out_main);
}
